// GatedDeltaNet_7181185319143
// MI455X (gfx1250) — matrix operand order repaired and checked by reading, not compiled, not run on hardware
//
#include <hip/hip_runtime.h>
#include <hip/hip_bf16.h>
#include <math.h>

// ---------------- problem constants ----------------
static constexpr int B_   = 2;
static constexpr int T_   = 1024;
static constexpr int H_   = 8;
static constexpr int HD_  = 64;
static constexpr int HID_ = 512;
static constexpr int KD_  = 512;      // H_*HD_
static constexpr int M_   = B_ * T_;  // 2048 rows

// ---------------- WMMA types ----------------
typedef __attribute__((ext_vector_type(16))) _Float16 v16h;
typedef __attribute__((ext_vector_type(8)))  float    v8f;

union AFrag { v16h v; _Float16 e[16]; };
union CFrag { v8f  v; float    e[8];  };

__device__ __forceinline__ float silu_f(float y) { return y / (1.f + expf(-y)); }

// =====================================================================
// f32 -> f16 conversion (one-time; makes GEMM tiles pure byte copies so
// the async global->LDS data mover path can be used)
// =====================================================================
__global__ __launch_bounds__(256) void cvt_f32_f16(
    const float* __restrict__ s, _Float16* __restrict__ d, int n)
{
    const int i = (blockIdx.x * 256 + threadIdx.x) * 4;
    if (i + 3 < n) {
        const float4 v = *(const float4*)(s + i);
        d[i]     = (_Float16)v.x;
        d[i + 1] = (_Float16)v.y;
        d[i + 2] = (_Float16)v.z;
        d[i + 3] = (_Float16)v.w;
    }
}

// =====================================================================
// WMMA GEMM (f16 in, f32 out):  C[M,N] = A[M,K] * Bw[N,K]^T
// block = 256 threads (8 waves); block tile 64(M) x 128(N); K-step 64.
// Double-buffered LDS tiles staged with GLOBAL_LOAD_ASYNC_TO_LDS_B128:
// next tile's 6 copies/wave overlap current tile's 8 WMMAs; partial
// s_wait_asynccnt 6 retires only the current tile (async ops complete
// in order per wave). Each wave owns a 32x32 patch:
// 2 A-frags x 2 B-frags x 2 K-halves = 8 v_wmma_f32_16x16x32_f16 / step.
// =====================================================================
__global__ __launch_bounds__(256) void wmma_gemm_f16(
    const _Float16* __restrict__ A, const _Float16* __restrict__ Bw,
    float* __restrict__ C, int M, int N, int K)
{
    __shared__ _Float16 As[2][64 * 64];     // 2 x 8 KB  (row stride 128 B)
    __shared__ _Float16 Bs[2][128 * 64];    // 2 x 16 KB (row stride 128 B)

    const int tid   = threadIdx.x;
    const int lane  = tid & 31;
    const int wid   = tid >> 5;
    const int m0    = blockIdx.y * 64;
    const int n0    = blockIdx.x * 128;
    const int wm    = wid & 1;           // 2 M-groups of 32 rows
    const int wn    = wid >> 1;          // 4 N-groups of 32 cols
    const int mlane = lane & 15;
    const int hi    = lane >> 4;         // 0 or 1

    // issue this thread's 6 async 16B-chunk copies for one K-step tile
    auto issue_tile = [&](int buf, int k0) {
        const unsigned ldsA = (unsigned)(uintptr_t)(void*)&As[buf][0];
        const unsigned ldsB = (unsigned)(uintptr_t)(void*)&Bs[buf][0];
#pragma unroll
        for (int c = 0; c < 2; ++c) {               // A: 512 chunks
            const int ci  = tid + c * 256;
            const int row = ci >> 3;                // 8 chunks per 128-B row
            const int cb  = (ci & 7) * 16;          // byte column
            const unsigned la = ldsA + row * 128 + cb;
            const unsigned long long ga =
                (unsigned long long)(uintptr_t)(A + (size_t)(m0 + row) * K + k0 + (cb >> 1));
            asm volatile("global_load_async_to_lds_b128 %0, %1, off"
                         :: "v"(la), "v"(ga) : "memory");
        }
#pragma unroll
        for (int c = 0; c < 4; ++c) {               // B: 1024 chunks
            const int ci  = tid + c * 256;
            const int row = ci >> 3;
            const int cb  = (ci & 7) * 16;
            const unsigned la = ldsB + row * 128 + cb;
            const unsigned long long ga =
                (unsigned long long)(uintptr_t)(Bw + (size_t)(n0 + row) * K + k0 + (cb >> 1));
            asm volatile("global_load_async_to_lds_b128 %0, %1, off"
                         :: "v"(la), "v"(ga) : "memory");
        }
    };

    CFrag acc[2][2];
#pragma unroll
    for (int mi = 0; mi < 2; ++mi)
#pragma unroll
        for (int ni = 0; ni < 2; ++ni)
#pragma unroll
            for (int r = 0; r < 8; ++r) acc[mi][ni].e[r] = 0.f;

    const int nsteps = K >> 6;           // K / 64
    issue_tile(0, 0);                    // prologue

    for (int s = 0; s < nsteps; ++s) {
        const int buf = s & 1;
        if (s + 1 < nsteps) {
            issue_tile(buf ^ 1, (s + 1) * 64);      // overlap with compute
            asm volatile("s_wait_asynccnt 0x6" ::: "memory");  // current tile done
        } else {
            asm volatile("s_wait_asynccnt 0x0" ::: "memory");
        }
        __syncthreads();                             // tile visible block-wide

        const _Float16* __restrict__ as = &As[buf][0];
        const _Float16* __restrict__ bs = &Bs[buf][0];
#pragma unroll
        for (int kh = 0; kh < 2; ++kh) {             // 2 K-halves of 32
            AFrag a[2], b[2];
#pragma unroll
            for (int mi = 0; mi < 2; ++mi) {
                const int arow = wm * 32 + mi * 16 + mlane;
#pragma unroll
                for (int j = 0; j < 16; ++j) {
                    const int vv = j >> 1, pos = j & 1;
                    const int kk = ((vv >= 4) ? 16 : 0) + hi * 8 + 2 * (vv & 3) + pos;
                    a[mi].e[j] = as[arow * 64 + kh * 32 + kk];
                }
            }
#pragma unroll
            for (int ni = 0; ni < 2; ++ni) {
                const int brow = wn * 32 + ni * 16 + mlane;
#pragma unroll
                for (int j = 0; j < 16; ++j)
                    b[ni].e[j] = bs[brow * 64 + kh * 32 + ((j >= 8) ? 16 : 0) + hi * 8 + (j & 7)];
            }
#pragma unroll
            for (int mi = 0; mi < 2; ++mi)
#pragma unroll
                for (int ni = 0; ni < 2; ++ni)
                    acc[mi][ni].v = __builtin_amdgcn_wmma_f32_16x16x32_f16(
                        false, a[mi].v, false, b[ni].v,
                        (short)0, acc[mi][ni].v, false, false);
        }
        __syncthreads();   // all reads of buf done before it is re-filled
    }

    // ---- write D (VGPR r: lanes0-15 M=r, lanes16-31 M=r+8) ----
#pragma unroll
    for (int mi = 0; mi < 2; ++mi)
#pragma unroll
        for (int ni = 0; ni < 2; ++ni) {
#pragma unroll
            for (int r = 0; r < 8; ++r) {
                const int orow = m0 + wm * 32 + mi * 16 + hi * 8 + r;
                const int ocol = n0 + wn * 32 + ni * 16 + mlane;
                C[(size_t)orow * N + ocol] = acc[mi][ni].e[r];
            }
        }
}

// =====================================================================
// small GEMM: Y[M,8] = X[M,K] * W[8,K]^T   (beta / gate-k projections)
// =====================================================================
__global__ __launch_bounds__(256) void small_gemm_n8(
    const float* __restrict__ X, const float* __restrict__ W,
    float* __restrict__ Y, int K)
{
    const int tid  = threadIdx.x;
    const int rloc = tid >> 3;                 // 32 rows per block
    const int n    = tid & 7;
    const int m    = blockIdx.x * 32 + rloc;
    const float* x = X + (size_t)m * K;
    const float* w = W + (size_t)n * K;
    float acc = 0.f;
    for (int k = 0; k < K; ++k) acc += x[k] * w[k];
    Y[(size_t)m * 8 + n] = acc;
}

// =====================================================================
// prep: causal depthwise conv(4)+SiLU for q/k/v, RoPE + L2 norm for q/k,
// beta = sigmoid, alpha = exp(-A*softplus). One block per (b,t).
// =====================================================================
__global__ __launch_bounds__(256) void prep_kernel(
    const float* __restrict__ qpre, const float* __restrict__ kpre,
    const float* __restrict__ vpre,
    const float* __restrict__ cqw, const float* __restrict__ cqb,
    const float* __restrict__ ckw, const float* __restrict__ ckb,
    const float* __restrict__ cvw, const float* __restrict__ cvb,
    const float* __restrict__ bpre, const float* __restrict__ gpre,
    const float* __restrict__ bb,   const float* __restrict__ bgk,
    const float* __restrict__ A_log,const float* __restrict__ dt_bias,
    float* __restrict__ qrot, float* __restrict__ krot,
    float* __restrict__ vconv, float* __restrict__ beta,
    float* __restrict__ alpha)
{
    const int bt  = blockIdx.x;          // b*T + t
    const int t   = bt % T_;
    const int tid = threadIdx.x;

    __shared__ float qs[KD_], ks[KD_], vs[KD_];

    for (int c = tid; c < KD_; c += 256) {
        float yq = cqb[c], yk = ckb[c], yv = cvb[c];
#pragma unroll
        for (int i = 0; i < 4; ++i) {
            const int tt = t - 3 + i;
            if (tt >= 0) {
                const size_t off = (size_t)(bt - t + tt) * KD_ + c;
                yq += cqw[c * 4 + i] * qpre[off];
                yk += ckw[c * 4 + i] * kpre[off];
                yv += cvw[c * 4 + i] * vpre[off];
            }
        }
        qs[c] = silu_f(yq); ks[c] = silu_f(yk); vs[c] = silu_f(yv);
    }
    __syncthreads();

    const int lane = tid & 31;
    const int h    = tid >> 5;           // wave == head (wave32)

    // RoPE angles: pair i uses freq index (2i)%32
    const float j_   = (float)((2 * lane) & 31);
    const float ar   = j_ * (1.f / 32.f);
    const float base = 10000.f;
    const float sbase = base * powf(32.f, 64.f / 62.f);
    const float invf = powf(sbase, -ar);
    const float fe   = powf(base, -ar);
    const float wl   = 6.28318530718f / fe;
    const float ramp = fminf(fmaxf((wl - 1.f) * (1.f / 31.f), 0.f), 1.f);
    const float spf  = 1.f + 31.f * ramp;
    const float fr   = ((float)t / spf) * invf;
    const float cv = cosf(fr), sv = sinf(fr);

    const size_t obase = (size_t)bt * KD_ + h * HD_;
    {   // q : rotate + l2norm (wave reduction)
        const float x1 = qs[h * HD_ + 2 * lane], x2 = qs[h * HD_ + 2 * lane + 1];
        float r1 = x1 * cv - x2 * sv, r2 = x1 * sv + x2 * cv;
        float ss = r1 * r1 + r2 * r2;
        for (int m = 16; m >= 1; m >>= 1) ss += __shfl_xor(ss, m, 32);
        const float sc = 1.f / fmaxf(sqrtf(ss), 1e-12f);
        qrot[obase + lane] = r1 * sc;  qrot[obase + 32 + lane] = r2 * sc;
    }
    {   // k
        const float x1 = ks[h * HD_ + 2 * lane], x2 = ks[h * HD_ + 2 * lane + 1];
        float r1 = x1 * cv - x2 * sv, r2 = x1 * sv + x2 * cv;
        float ss = r1 * r1 + r2 * r2;
        for (int m = 16; m >= 1; m >>= 1) ss += __shfl_xor(ss, m, 32);
        const float sc = 1.f / fmaxf(sqrtf(ss), 1e-12f);
        krot[obase + lane] = r1 * sc;  krot[obase + 32 + lane] = r2 * sc;
    }
    // v passthrough
    vconv[(size_t)bt * KD_ + tid]       = vs[tid];
    vconv[(size_t)bt * KD_ + 256 + tid] = vs[256 + tid];

    if (tid < H_) {
        const int hh = tid;
        const float bv = bpre[(size_t)bt * H_ + hh] + bb[hh];
        beta[(size_t)bt * H_ + hh] = 1.f / (1.f + expf(-bv));
        const float gv = gpre[(size_t)bt * H_ + hh] + bgk[hh] + dt_bias[hh];
        const float sp = (gv > 20.f) ? gv : log1pf(expf(gv));
        alpha[(size_t)bt * H_ + hh] = expf(-expf(A_log[hh]) * sp);
    }
}

// =====================================================================
// gated delta-rule scan. One block per (b,h). 64x64 f32 state held in
// registers: thread (gq = tid>>6) owns rows [16*gq,16*gq+16), col e = tid&63.
// 3 barriers / timestep; LDS partial reductions for k.S and q.S.
// =====================================================================
__global__ __launch_bounds__(256) void scan_kernel(
    const float* __restrict__ qrot, const float* __restrict__ krot,
    const float* __restrict__ vconv, const float* __restrict__ beta,
    const float* __restrict__ alpha, float* __restrict__ ocore)
{
    const int bh = blockIdx.x;
    const int b  = bh / H_, h = bh % H_;
    const int tid = threadIdx.x;
    const int e   = tid & 63;
    const int gq  = tid >> 6;     // 0..3

    float S[16];
#pragma unroll
    for (int r = 0; r < 16; ++r) S[r] = 0.f;

    __shared__ float kv[HD_], qv[HD_], vv[HD_];
    __shared__ float red1[4][HD_], red2[4][HD_];

    for (int t = 0; t < T_; ++t) {
        const size_t base = ((size_t)(b * T_ + t) * H_ + h) * HD_;
        if (tid < HD_) {
            kv[tid] = krot[base + tid];
            qv[tid] = qrot[base + tid];
            vv[tid] = vconv[base + tid];
        }
        const float at = alpha[(size_t)(b * T_ + t) * H_ + h];
        const float bt = beta [(size_t)(b * T_ + t) * H_ + h];
        __syncthreads();                       // kv/qv/vv ready

        float p = 0.f;
#pragma unroll
        for (int r = 0; r < 16; ++r) p += kv[gq * 16 + r] * S[r];
        red1[gq][e] = p;
        __syncthreads();                       // partials ready

        const float kS = red1[0][e] + red1[1][e] + red1[2][e] + red1[3][e];
        const float bvv = bt * vv[e];
        float p2 = 0.f;
#pragma unroll
        for (int r = 0; r < 16; ++r) {
            const float kd = kv[gq * 16 + r];
            S[r] = at * (S[r] - bt * kd * kS) + bvv * kd;
            p2  += qv[gq * 16 + r] * S[r];
        }
        red2[gq][e] = p2;
        __syncthreads();                       // output partials ready
        if (gq == 0)
            ocore[base + e] = red2[0][e] + red2[1][e] + red2[2][e] + red2[3][e];
    }
}

// =====================================================================
// post: o = ocore + D*v; RMS-norm (per head); att = g * silu(o_n)
// wave per head, lane handles dims (lane, lane+32). Emits f16 so the
// output projection consumes it directly.
// =====================================================================
__global__ __launch_bounds__(256) void post_kernel(
    const float* __restrict__ ocore, const float* __restrict__ vconv,
    const float* __restrict__ gbuf,  const float* __restrict__ Dp,
    const float* __restrict__ onw,   _Float16* __restrict__ atth)
{
    const int bt  = blockIdx.x;
    const int tid = threadIdx.x;
    const int lane = tid & 31, h = tid >> 5;
    const size_t base = (size_t)bt * KD_ + h * HD_;
    const float dv = Dp[h];
    const float o1 = ocore[base + lane]      + dv * vconv[base + lane];
    const float o2 = ocore[base + 32 + lane] + dv * vconv[base + 32 + lane];
    float ss = o1 * o1 + o2 * o2;
    for (int m = 16; m >= 1; m >>= 1) ss += __shfl_xor(ss, m, 32);
    const float inv = rsqrtf(ss * (1.f / 64.f) + 1e-6f);
    const float n1 = o1 * inv * onw[lane];
    const float n2 = o2 * inv * onw[32 + lane];
    atth[base + lane]      = (_Float16)(gbuf[base + lane]      * silu_f(n1));
    atth[base + 32 + lane] = (_Float16)(gbuf[base + 32 + lane] * silu_f(n2));
}

// =====================================================================
// host launcher
// =====================================================================
extern "C" void kernel_launch(void* const* d_in, const int* in_sizes, int n_in,
                              void* d_out, int out_size, void* d_ws, size_t ws_size,
                              hipStream_t stream)
{
    (void)in_sizes; (void)n_in; (void)out_size; (void)ws_size;

    const float* x     = (const float*)d_in[0];
    const float* Wq    = (const float*)d_in[1];
    const float* Wk    = (const float*)d_in[2];
    const float* Wv    = (const float*)d_in[3];
    const float* Wg    = (const float*)d_in[4];
    const float* Wo    = (const float*)d_in[5];
    const float* Wb    = (const float*)d_in[6];
    const float* bb    = (const float*)d_in[7];
    const float* Wgk   = (const float*)d_in[8];
    const float* bgk   = (const float*)d_in[9];
    const float* cqw   = (const float*)d_in[10];
    const float* cqb   = (const float*)d_in[11];
    const float* ckw   = (const float*)d_in[12];
    const float* ckb   = (const float*)d_in[13];
    const float* cvw   = (const float*)d_in[14];
    const float* cvb   = (const float*)d_in[15];
    const float* A_log = (const float*)d_in[16];
    const float* Dp    = (const float*)d_in[17];
    const float* dtb   = (const float*)d_in[18];
    const float* onw   = (const float*)d_in[19];
    float* out = (float*)d_out;

    char* ws = (char*)d_ws;
    const size_t MB4 = (size_t)M_ * KD_ * sizeof(float);   // 4 MiB
    float* qpre  = (float*)(ws + 0 * MB4);
    float* kpre  = (float*)(ws + 1 * MB4);
    float* vpre  = (float*)(ws + 2 * MB4);
    float* gbuf  = (float*)(ws + 3 * MB4);
    float* qrot  = (float*)(ws + 4 * MB4);
    float* krot  = (float*)(ws + 5 * MB4);
    float* vconv = (float*)(ws + 6 * MB4);
    float* ocore = kpre;                                    // reuse (kpre dead after prep)

    char*  small = ws + 7 * MB4;
    const size_t SB = (size_t)M_ * H_ * sizeof(float);      // 64 KiB
    float* bpre  = (float*)(small + 0 * SB);
    float* gpre  = (float*)(small + 1 * SB);
    float* beta  = (float*)(small + 2 * SB);
    float* alpha = (float*)(small + 3 * SB);

    // f16 staging buffers (so GEMM tiles are raw byte copies -> async LDS)
    _Float16* xh   = (_Float16*)(small + 4 * SB);           // 2 MiB
    _Float16* Wqh  = xh  + (size_t)M_ * HID_;
    _Float16* Wkh  = Wqh + (size_t)KD_ * HID_;
    _Float16* Wvh  = Wkh + (size_t)KD_ * HID_;
    _Float16* Wgh  = Wvh + (size_t)KD_ * HID_;
    _Float16* Woh  = Wgh + (size_t)KD_ * HID_;
    _Float16* atth = Woh + (size_t)HID_ * KD_;              // 2 MiB

    const dim3 blk(256);

    // 0) one-time f16 conversions
    cvt_f32_f16<<<dim3((M_ * HID_) / 1024), blk, 0, stream>>>(x,  xh,  M_ * HID_);
    cvt_f32_f16<<<dim3((KD_ * HID_) / 1024), blk, 0, stream>>>(Wq, Wqh, KD_ * HID_);
    cvt_f32_f16<<<dim3((KD_ * HID_) / 1024), blk, 0, stream>>>(Wk, Wkh, KD_ * HID_);
    cvt_f32_f16<<<dim3((KD_ * HID_) / 1024), blk, 0, stream>>>(Wv, Wvh, KD_ * HID_);
    cvt_f32_f16<<<dim3((KD_ * HID_) / 1024), blk, 0, stream>>>(Wg, Wgh, KD_ * HID_);
    cvt_f32_f16<<<dim3((HID_ * KD_) / 1024), blk, 0, stream>>>(Wo, Woh, HID_ * KD_);

    // 1) input projections (WMMA, double-buffered async-LDS staging)
    const dim3 gemmGrid(KD_ / 128, M_ / 64);   // (4, 32)
    wmma_gemm_f16<<<gemmGrid, blk, 0, stream>>>(xh, Wqh, qpre, M_, KD_, HID_);
    wmma_gemm_f16<<<gemmGrid, blk, 0, stream>>>(xh, Wkh, kpre, M_, KD_, HID_);
    wmma_gemm_f16<<<gemmGrid, blk, 0, stream>>>(xh, Wvh, vpre, M_, KD_, HID_);
    wmma_gemm_f16<<<gemmGrid, blk, 0, stream>>>(xh, Wgh, gbuf, M_, KD_, HID_);

    // 2) beta / gate-k pre-activations
    small_gemm_n8<<<dim3(M_ / 32), blk, 0, stream>>>(x, Wb,  bpre, HID_);
    small_gemm_n8<<<dim3(M_ / 32), blk, 0, stream>>>(x, Wgk, gpre, HID_);

    // 3) conv + SiLU + RoPE + L2norm + beta/alpha
    prep_kernel<<<dim3(B_ * T_), blk, 0, stream>>>(
        qpre, kpre, vpre, cqw, cqb, ckw, ckb, cvw, cvb,
        bpre, gpre, bb, bgk, A_log, dtb,
        qrot, krot, vconv, beta, alpha);

    // 4) sequential gated delta scan (one block per (b,h))
    scan_kernel<<<dim3(B_ * H_), blk, 0, stream>>>(qrot, krot, vconv, beta, alpha, ocore);

    // 5) +D*v, RMS norm, gating -> f16
    post_kernel<<<dim3(B_ * T_), blk, 0, stream>>>(ocore, vconv, gbuf, Dp, onw, atth);

    // 6) output projection (WMMA) -> d_out
    wmma_gemm_f16<<<dim3(HID_ / 128, M_ / 64), blk, 0, stream>>>(atth, Woh, out, M_, HID_, KD_);
}
